// SlidingWindowAttention_35038343201012
// MI455X (gfx1250) — matrix operand order repaired and checked by reading, not compiled, not run on hardware
//
#include <hip/hip_runtime.h>

typedef __attribute__((ext_vector_type(16))) _Float16 v16h;
typedef __attribute__((ext_vector_type(8)))  _Float16 v8h;
typedef __attribute__((ext_vector_type(8)))  float    v8f;

#define WINDOW 128
#define DHEAD  64

static __device__ __forceinline__ v8f wmma16x16x32(v16h a, v16h b, v8f c) {
  // 8 args: (neg_a, A, neg_b, B, c_mod, C, reuse_a, reuse_b)
  return __builtin_amdgcn_wmma_f32_16x16x32_f16(false, a, false, b, (short)0, c,
                                                false, false);
}

static __device__ __forceinline__ v16h lds_load_v16h(const _Float16* p) {
  v8h lo = *(const v8h*)(p);
  v8h hh = *(const v8h*)(p + 8);
  return __builtin_shufflevector(lo, hh, 0, 1, 2, 3, 4, 5, 6, 7,
                                 8, 9, 10, 11, 12, 13, 14, 15);
}

// One workgroup handles one (batch*head, query-block) pair:
//   scores S[128 x 256] = Qblk (128x64) * [Kprev;Kcur]^T, masked softmax,
//   O[128 x 64] = P * [Vprev;Vcur].
// 8 waves; wave w owns query rows [16w, 16w+16).
__global__ __launch_bounds__(256, 1)
void swa_fused_kernel(const float* __restrict__ Q, const float* __restrict__ K,
                      const float* __restrict__ V, float* __restrict__ O,
                      int L, int nb) {
  __shared__ __align__(16) _Float16 Ksm[2 * WINDOW * DHEAD]; // [256][64], reused as P scratch
  __shared__ __align__(16) _Float16 Vt[DHEAD * 2 * WINDOW];  // [64][256] (transposed V)

  const int tid  = threadIdx.x;
  const int wv   = tid >> 5;   // wave id 0..7
  const int lane = tid & 31;
  const int ln   = lane & 15;  // A-row / B-col / D-col within a 16x16 tile
  const int hi   = lane >> 4;  // half-wave selector
  const int kb   = hi * 8;     // A-operand K-group base

  const int bh  = blockIdx.x / nb;
  const int blk = blockIdx.x % nb;

  const size_t headoff = (size_t)bh * (size_t)L * DHEAD;
  const float* Qh = Q + headoff;
  const float* Kh = K + headoff;
  const float* Vh = V + headoff;
  float*       Oh = O + headoff;

  // ---- Stage K,V (prev block + current block) into LDS as f16 ----
  // Coalesced mapping: per iteration a wave's 32 lanes cover one contiguous
  // 512B span (2 rows x 64 cols). Thread handles rows (tid>>4)+16*i at
  // columns (tid&15)*4 .. +3.
  {
    const int rbase = tid >> 4;        // 0..15
    const int c4    = (tid & 15) * 4;  // 0,4,...,60
    #pragma unroll
    for (int i = 0; i < 16; ++i) {
      const int kv  = rbase + 16 * i;           // 0..255 local kv row
      const int gkv = (blk - 1) * WINDOW + kv;  // global kv position
      float4 kf, vf;
      if (blk == 0 && kv < WINDOW) {
        kf = make_float4(0.f, 0.f, 0.f, 0.f);
        vf = kf;
      } else {
        kf = *(const float4*)(Kh + (size_t)gkv * DHEAD + c4);
        vf = *(const float4*)(Vh + (size_t)gkv * DHEAD + c4);
      }
      Ksm[kv * DHEAD + c4 + 0] = (_Float16)kf.x;
      Ksm[kv * DHEAD + c4 + 1] = (_Float16)kf.y;
      Ksm[kv * DHEAD + c4 + 2] = (_Float16)kf.z;
      Ksm[kv * DHEAD + c4 + 3] = (_Float16)kf.w;
      Vt[(c4 + 0) * (2 * WINDOW) + kv] = (_Float16)vf.x;
      Vt[(c4 + 1) * (2 * WINDOW) + kv] = (_Float16)vf.y;
      Vt[(c4 + 2) * (2 * WINDOW) + kv] = (_Float16)vf.z;
      Vt[(c4 + 3) * (2 * WINDOW) + kv] = (_Float16)vf.w;
    }
  }
  __syncthreads();

  // ---- Load Q rows for this wave as two 16x32 A-operands (pre-scaled) ----
  const float scale = 0.125f; // 1/sqrt(64)
  const float* qrow = Qh + (size_t)(blk * WINDOW + wv * 16 + ln) * DHEAD;
  v16h aQ[2];
  #pragma unroll
  for (int s = 0; s < 2; ++s) {
    #pragma unroll
    for (int e = 0; e < 8; ++e) {
      aQ[s][e]     = (_Float16)(qrow[s * 32 + hi * 16 + e] * scale);
      aQ[s][e + 8] = (_Float16)(qrow[s * 32 + hi * 16 + 8 + e] * scale);
    }
  }

  // ---- Scores: 16 kv-tiles x (2 k-steps) WMMAs ----
  v8f acc[16];
  #pragma unroll
  for (int t = 0; t < 16; ++t) {
    v8f c = {};
    #pragma unroll
    for (int s = 0; s < 2; ++s) {
      // B element [k=d][n=kv]: lane holds kv = t*16+ln, d = s*32 + hi*16 + e
      v16h b = lds_load_v16h(&Ksm[(t * 16 + ln) * DHEAD + s * 32 + hi * 16]);
      c = wmma16x16x32(aQ[s], b, c);
    }
    acc[t] = c;
  }

  // ---- Masked softmax over 256 columns per query row ----
  float rmax[8], rsum[8], rinv[8];
  #pragma unroll
  for (int r = 0; r < 8; ++r) rmax[r] = -3.0e38f;
  #pragma unroll
  for (int t = 0; t < 16; ++t) {
    const int j = t * 16 + ln; // kv column (0..255)
    #pragma unroll
    for (int r = 0; r < 8; ++r) {
      const int qi = wv * 16 + hi * 8 + r; // query index in block (0..127)
      const bool valid =
          (j > qi) && (j <= qi + WINDOW) && ((blk > 0) || (j >= WINDOW));
      float sv = valid ? acc[t][r] : -__builtin_inff();
      acc[t][r] = sv;
      rmax[r] = fmaxf(rmax[r], sv);
    }
  }
  #pragma unroll
  for (int m = 1; m <= 8; m <<= 1) {
    #pragma unroll
    for (int r = 0; r < 8; ++r)
      rmax[r] = fmaxf(rmax[r], __shfl_xor(rmax[r], m, 32));
  }
  #pragma unroll
  for (int r = 0; r < 8; ++r) rsum[r] = 0.0f;
  #pragma unroll
  for (int t = 0; t < 16; ++t) {
    #pragma unroll
    for (int r = 0; r < 8; ++r) {
      float p = __expf(acc[t][r] - rmax[r]); // exp(-inf)=0 handles mask
      acc[t][r] = p;
      rsum[r] += p;
    }
  }
  #pragma unroll
  for (int m = 1; m <= 8; m <<= 1) {
    #pragma unroll
    for (int r = 0; r < 8; ++r)
      rsum[r] += __shfl_xor(rsum[r], m, 32);
  }
  #pragma unroll
  for (int r = 0; r < 8; ++r) rinv[r] = 1.0f / rsum[r];

  __syncthreads(); // all waves done reading Ksm -> reuse it as P scratch

  // ---- O = P * V, via per-wave LDS transpose of P (D-layout -> A-layout) ----
  _Float16* scr = Ksm + wv * (16 * 128); // wave-private 16x128 f16, row-major
  v8f oacc[4];
  #pragma unroll
  for (int t = 0; t < 4; ++t) { v8f z = {}; oacc[t] = z; }

  #pragma unroll
  for (int h = 0; h < 2; ++h) {
    // store normalized P half (columns h*128..h*128+127) to scratch
    #pragma unroll
    for (int t2 = 0; t2 < 8; ++t2) {
      const int t = h * 8 + t2;
      #pragma unroll
      for (int r = 0; r < 8; ++r) {
        scr[(hi * 8 + r) * 128 + t2 * 16 + ln] =
            (_Float16)(acc[t][r] * rinv[r]);
      }
    }
    // P(16x128) * Vhalf(128x64): 4 k-steps x 4 d-tiles
    #pragma unroll
    for (int s = 0; s < 4; ++s) {
      const _Float16* p0 = scr + ln * 128 + s * 32 + hi * 16;
      v8h lo = *(const v8h*)(p0);
      v8h hh = *(const v8h*)(p0 + 8);
      v16h aP = __builtin_shufflevector(lo, hh, 0, 1, 2, 3, 4, 5, 6, 7,
                                        8, 9, 10, 11, 12, 13, 14, 15);
      #pragma unroll
      for (int t = 0; t < 4; ++t) {
        // B element [k=kv][n=d]: Vt[d][h*128 + s*32 + hi*16 + e]
        v16h bV = lds_load_v16h(
            &Vt[(t * 16 + ln) * (2 * WINDOW) + h * WINDOW + s * 32 + hi * 16]);
        oacc[t] = wmma16x16x32(aP, bV, oacc[t]);
      }
    }
  }

  // ---- Epilogue: write O (f32) from D-layout accumulators ----
  float* orow0 = Oh + (size_t)(blk * WINDOW + wv * 16) * DHEAD;
  #pragma unroll
  for (int t = 0; t < 4; ++t) {
    #pragma unroll
    for (int r = 0; r < 8; ++r) {
      orow0[(size_t)(hi * 8 + r) * DHEAD + t * 16 + ln] = oacc[t][r];
    }
  }
}

extern "C" void kernel_launch(void* const* d_in, const int* in_sizes, int n_in,
                              void* d_out, int out_size, void* d_ws, size_t ws_size,
                              hipStream_t stream) {
  (void)in_sizes; (void)n_in; (void)d_ws; (void)ws_size; (void)out_size;
  const float* Q = (const float*)d_in[0];
  const float* K = (const float*)d_in[1];
  const float* V = (const float*)d_in[2];
  float*       O = (float*)d_out;

  const int B = 2, H = 16, L = 8192;
  const int nb = L / WINDOW; // 64
  dim3 grid(B * H * nb);     // 2048 workgroups
  dim3 block(256);           // 8 wave32 waves
  hipLaunchKernelGGL(swa_fused_kernel, grid, block, 0, stream, Q, K, V, O, L, nb);
}
